// SelfAttention_87771951661553
// MI455X (gfx1250) — matrix operand order repaired and checked by reading, not compiled, not run on hardware
//
#include <hip/hip_runtime.h>
#include <hip/hip_bf16.h>

typedef __attribute__((ext_vector_type(16))) __bf16 v16bf;
typedef __attribute__((ext_vector_type(8)))  __bf16 v8bf;
typedef __attribute__((ext_vector_type(8)))  float  v8f;
typedef __attribute__((ext_vector_type(4)))  int    v4i;

#define E_DIM 1024
#define N_HEAD 16
#define HEAD_DIM 64
#define SEQ_T 2048
#define N3 3072
#define LOG2E 1.4426950408889634f

#if defined(__has_builtin)
#if __has_builtin(__builtin_amdgcn_global_load_async_to_lds_b128)
#define USE_ASYNC_LDS 1
#endif
#endif

#ifdef USE_ASYNC_LDS
#define AS1P __attribute__((address_space(1))) v4i*
#define AS3P __attribute__((address_space(3))) v4i*
#define ASYNC_CP16(gp, lp)                                                    \
  __builtin_amdgcn_global_load_async_to_lds_b128((AS1P)(gp), (AS3P)(lp), 0, 0)
#define ASYNC_WAIT() asm volatile("s_wait_asynccnt 0" ::: "memory")
#endif

__device__ __forceinline__ v16bf cat16(v8bf lo, v8bf hi) {
  v16bf r;
#pragma unroll
  for (int i = 0; i < 8; ++i) { r[i] = lo[i]; r[i + 8] = hi[i]; }
  return r;
}

// ---- DPP 16-lane butterfly reductions (VALU pipe, no LDS traffic) ----
template <int CTRL>
__device__ __forceinline__ float dpp_mov(float x) {
  int v = __builtin_amdgcn_update_dpp(0, __builtin_bit_cast(int, x), CTRL,
                                      0xF, 0xF, true);
  return __builtin_bit_cast(float, v);
}
__device__ __forceinline__ float row16_max(float x) {
  x = fmaxf(x, dpp_mov<0xB1>(x));   // quad_perm {1,0,3,2}
  x = fmaxf(x, dpp_mov<0x4E>(x));   // quad_perm {2,3,0,1}
  x = fmaxf(x, dpp_mov<0x141>(x));  // row_half_mirror
  x = fmaxf(x, dpp_mov<0x140>(x));  // row_mirror
  return x;
}
__device__ __forceinline__ float row16_sum(float x) {
  x += dpp_mov<0xB1>(x);
  x += dpp_mov<0x4E>(x);
  x += dpp_mov<0x141>(x);
  x += dpp_mov<0x140>(x);
  return x;
}

// ---------------- converts ----------------
__global__ void cvt_f32_to_bf16(const float* __restrict__ in,
                                __bf16* __restrict__ out, int n) {
  int i = blockIdx.x * blockDim.x + threadIdx.x;
  if (i < n) out[i] = (__bf16)in[i];
}

// in: [K][N] f32 row-major -> out: [N][K] bf16 ("transposed" weights)
__global__ void cvt_transpose_bf16(const float* __restrict__ in,
                                   __bf16* __restrict__ out, int K, int N) {
  int i = blockIdx.x * blockDim.x + threadIdx.x;
  if (i < K * N) {
    int n = i / K, k = i % K;
    out[i] = (__bf16)in[(size_t)k * N + n];
  }
}

// ---------------- generic bf16 GEMM ----------------
// A: [M][K] bf16 row-major, Bt: [N][K] bf16 (pre-transposed), bias: [N] f32
// MODE 0: C = A*B + bias, f32 [M][N]
// MODE 1: scatter into Q/K/V [B][H][T][HEAD_DIM] bf16 (N == 3*E)
template <int MODE>
__global__ __launch_bounds__(256) void gemm_bf16(
    const __bf16* __restrict__ A, const __bf16* __restrict__ Bt,
    const float* __restrict__ bias, float* __restrict__ C,
    __bf16* __restrict__ qo, __bf16* __restrict__ ko, __bf16* __restrict__ vo,
    int M, int N, int K, int T) {
  constexpr int BM = 128, BN = 128, BK = 32;
  __shared__ __bf16 As[BM][BK];
  __shared__ __bf16 Bs[BN][BK];

  const int tid = threadIdx.x;
  const int wave = tid >> 5, lane = tid & 31;
  const int lm = lane & 15, lh = lane >> 4;
  const int tileM = blockIdx.y * BM, tileN = blockIdx.x * BN;
  const int waveM = (wave >> 1) * 32, waveN = (wave & 1) * 64;

  v8f acc[2][4];
#pragma unroll
  for (int i = 0; i < 2; ++i)
#pragma unroll
    for (int j = 0; j < 4; ++j)
#pragma unroll
      for (int r = 0; r < 8; ++r) acc[i][j][r] = 0.0f;

  const int crow = tid >> 1;            // 0..127
  const int ccol = (tid & 1) * 16;      // 0 or 16

  for (int kk = 0; kk < K; kk += BK) {
    // cooperative tile staging: 16 bf16 per thread per tile, contiguous
    {
      const __bf16* ap = &A[(size_t)(tileM + crow) * K + kk + ccol];
      const __bf16* bp = &Bt[(size_t)(tileN + crow) * K + kk + ccol];
#ifdef USE_ASYNC_LDS
      ASYNC_CP16(ap, &As[crow][ccol]);
      ASYNC_CP16(ap + 8, &As[crow][ccol + 8]);
      ASYNC_CP16(bp, &Bs[crow][ccol]);
      ASYNC_CP16(bp + 8, &Bs[crow][ccol + 8]);
#else
      *(v8bf*)&As[crow][ccol]     = *(const v8bf*)(ap);
      *(v8bf*)&As[crow][ccol + 8] = *(const v8bf*)(ap + 8);
      *(v8bf*)&Bs[crow][ccol]     = *(const v8bf*)(bp);
      *(v8bf*)&Bs[crow][ccol + 8] = *(const v8bf*)(bp + 8);
#endif
      if (kk + BK < K) {  // pull next tiles toward L2/L0
        __builtin_prefetch(ap + BK, 0, 1);
        __builtin_prefetch(bp + BK, 0, 1);
      }
#ifdef USE_ASYNC_LDS
      ASYNC_WAIT();
#endif
    }
    __syncthreads();

    v16bf afrag[2], bfrag[4];
#pragma unroll
    for (int i = 0; i < 2; ++i) {
      const int row = waveM + i * 16 + lm;
      afrag[i] = cat16(*(const v8bf*)&As[row][8 * lh],
                       *(const v8bf*)&As[row][16 + 8 * lh]);
    }
#pragma unroll
    for (int j = 0; j < 4; ++j) {
      const int col = waveN + j * 16 + lm;
      bfrag[j] = cat16(*(const v8bf*)&Bs[col][8 * lh],
                       *(const v8bf*)&Bs[col][16 + 8 * lh]);
    }
#pragma unroll
    for (int i = 0; i < 2; ++i)
#pragma unroll
      for (int j = 0; j < 4; ++j)
        acc[i][j] = __builtin_amdgcn_wmma_f32_16x16x32_bf16(
            false, afrag[i], false, bfrag[j], (short)0, acc[i][j], false, false);
    __syncthreads();
  }

  // epilogue
#pragma unroll
  for (int i = 0; i < 2; ++i) {
#pragma unroll
    for (int j = 0; j < 4; ++j) {
      const int n = tileN + waveN + j * 16 + lm;
      const float bv = bias[n];
#pragma unroll
      for (int r = 0; r < 8; ++r) {
        const int m = tileM + waveM + i * 16 + r + 8 * lh;
        const float val = acc[i][j][r] + bv;
        if (MODE == 0) {
          C[(size_t)m * N + n] = val;
        } else {
          const int jsel = n >> 10, rem = n & 1023;
          const int h = rem >> 6, d = rem & 63;
          const int t = m % T, b = m / T;
          __bf16* dst = (jsel == 0) ? qo : (jsel == 1) ? ko : vo;
          dst[((size_t)(b * N_HEAD + h) * T + t) * HEAD_DIM + d] = (__bf16)val;
        }
      }
    }
  }
}

// ---------------- flash attention ----------------
// Q,K,V: [B*H][T][64] bf16 ; O: [B][T][E] bf16
__global__ __launch_bounds__(128) void flash_attn(
    const __bf16* __restrict__ Q, const __bf16* __restrict__ Kg,
    const __bf16* __restrict__ V, __bf16* __restrict__ O, int T) {
  __shared__ __bf16 Ks[32][64];       // [key][d]
  __shared__ __bf16 Vt[64][40];       // [d][key], padded
  __shared__ __bf16 Pl[4][16][32];    // per-wave P staging

  const int tid = threadIdx.x;
  const int wave = tid >> 5, lane = tid & 31;
  const int lm = lane & 15, lh = lane >> 4;

  const int nqt = T / 64;
  const int qt = blockIdx.x % nqt;
  const int bh = blockIdx.x / nqt;          // b*N_HEAD + h
  const int h = bh % N_HEAD, b = bh / N_HEAD;

  const __bf16* Qb = Q + (size_t)bh * T * HEAD_DIM;
  const __bf16* Kb = Kg + (size_t)bh * T * HEAD_DIM;
  const __bf16* Vb = V + (size_t)bh * T * HEAD_DIM;

  const int q0 = qt * 64 + wave * 16;

  // Q fragments (held for the whole block)
  v16bf aq[2];
#pragma unroll
  for (int dc = 0; dc < 2; ++dc) {
    const __bf16* qp = &Qb[(size_t)(q0 + lm) * HEAD_DIM + dc * 32];
    aq[dc] = cat16(*(const v8bf*)(qp + 8 * lh),
                   *(const v8bf*)(qp + 16 + 8 * lh));
  }

  v8f o[4];
  float mrow[8], lrow[8];
#pragma unroll
  for (int t = 0; t < 4; ++t)
#pragma unroll
    for (int r = 0; r < 8; ++r) o[t][r] = 0.0f;
#pragma unroll
  for (int r = 0; r < 8; ++r) { mrow[r] = -3.0e38f; lrow[r] = 0.0f; }

  const float cscale = 0.125f * LOG2E;  // HEAD_DIM^-0.5 * log2(e)
  const int lkey = tid >> 2, lpart = (tid & 3) * 16;

  for (int kt = 0; kt < T / 32; ++kt) {
    // stage K tile (row-major, async if available) and V tile (transposed)
    {
      const __bf16* kp = &Kb[(size_t)(kt * 32 + lkey) * HEAD_DIM + lpart];
      const __bf16* vp = &Vb[(size_t)(kt * 32 + lkey) * HEAD_DIM + lpart];
#ifdef USE_ASYNC_LDS
      ASYNC_CP16(kp, &Ks[lkey][lpart]);
      ASYNC_CP16(kp + 8, &Ks[lkey][lpart + 8]);
#else
      *(v8bf*)&Ks[lkey][lpart]     = *(const v8bf*)(kp);
      *(v8bf*)&Ks[lkey][lpart + 8] = *(const v8bf*)(kp + 8);
#endif
      v8bf v0 = *(const v8bf*)(vp), v1 = *(const v8bf*)(vp + 8);
#pragma unroll
      for (int e = 0; e < 8; ++e) {
        Vt[lpart + e][lkey] = v0[e];
        Vt[lpart + 8 + e][lkey] = v1[e];
      }
      if (kt + 1 < T / 32) {
        __builtin_prefetch(kp + 32 * HEAD_DIM, 0, 1);
        __builtin_prefetch(vp + 32 * HEAD_DIM, 0, 1);
      }
#ifdef USE_ASYNC_LDS
      ASYNC_WAIT();
#endif
    }
    __syncthreads();

    // S = Q * K^T  (two 16-key subtiles)
    v8f s[2];
#pragma unroll
    for (int ks = 0; ks < 2; ++ks)
#pragma unroll
      for (int r = 0; r < 8; ++r) s[ks][r] = 0.0f;
#pragma unroll
    for (int ks = 0; ks < 2; ++ks) {
      const int key = ks * 16 + lm;
#pragma unroll
      for (int dc = 0; dc < 2; ++dc) {
        v16bf bk = cat16(*(const v8bf*)&Ks[key][dc * 32 + 8 * lh],
                         *(const v8bf*)&Ks[key][dc * 32 + 16 + 8 * lh]);
        s[ks] = __builtin_amdgcn_wmma_f32_16x16x32_bf16(
            false, aq[dc], false, bk, (short)0, s[ks], false, false);
      }
    }

    // online softmax (exp2 domain); rows r+8*lh uniform across 16-lane half
    float p0[8], p1[8];
#pragma unroll
    for (int r = 0; r < 8; ++r) {
      float s0 = s[0][r] * cscale, s1 = s[1][r] * cscale;
      const float mx = row16_max(fmaxf(s0, s1));
      const float mn = fmaxf(mrow[r], mx);
      const float alpha = exp2f(mrow[r] - mn);
      p0[r] = exp2f(s0 - mn);
      p1[r] = exp2f(s1 - mn);
      const float rs = row16_sum(p0[r] + p1[r]);
      lrow[r] = lrow[r] * alpha + rs;
      mrow[r] = mn;
#pragma unroll
      for (int t = 0; t < 4; ++t) o[t][r] *= alpha;
    }

    // C-layout -> A-layout via per-wave LDS staging
#pragma unroll
    for (int r = 0; r < 8; ++r) {
      Pl[wave][r + 8 * lh][lm] = (__bf16)p0[r];
      Pl[wave][r + 8 * lh][16 + lm] = (__bf16)p1[r];
    }
    asm volatile("s_wait_dscnt 0" ::: "memory");
    v16bf pa = cat16(*(const v8bf*)&Pl[wave][lm][8 * lh],
                     *(const v8bf*)&Pl[wave][lm][16 + 8 * lh]);

    // O += P * V
#pragma unroll
    for (int dt = 0; dt < 4; ++dt) {
      const int d = dt * 16 + lm;
      v16bf bv = cat16(*(const v8bf*)&Vt[d][8 * lh],
                       *(const v8bf*)&Vt[d][16 + 8 * lh]);
      o[dt] = __builtin_amdgcn_wmma_f32_16x16x32_bf16(
          false, pa, false, bv, (short)0, o[dt], false, false);
    }
    __syncthreads();
  }

  // normalize + write [B][T][E] bf16
#pragma unroll
  for (int r = 0; r < 8; ++r) {
    const float inv = 1.0f / lrow[r];
    const int q = q0 + r + 8 * lh;
#pragma unroll
    for (int dt = 0; dt < 4; ++dt) {
      const int col = h * HEAD_DIM + dt * 16 + lm;
      O[((size_t)b * T + q) * E_DIM + col] = (__bf16)(o[dt][r] * inv);
    }
  }
}

// ---------------- launcher ----------------
extern "C" void kernel_launch(void* const* d_in, const int* in_sizes, int n_in,
                              void* d_out, int out_size, void* d_ws, size_t ws_size,
                              hipStream_t stream) {
  const float* x      = (const float*)d_in[0];
  const float* w_qkv  = (const float*)d_in[1];
  const float* b_qkv  = (const float*)d_in[2];
  const float* w_proj = (const float*)d_in[3];
  const float* b_proj = (const float*)d_in[4];

  const int T = SEQ_T;                      // per reference
  const int M = in_sizes[0] / E_DIM;        // B*T rows

  char* w = (char*)d_ws;
  __bf16* x_bf    = (__bf16*)w; w += (size_t)M * E_DIM * 2;
  __bf16* wqkv_t  = (__bf16*)w; w += (size_t)N3 * E_DIM * 2;
  __bf16* wproj_t = (__bf16*)w; w += (size_t)E_DIM * E_DIM * 2;
  __bf16* Qb      = (__bf16*)w; w += (size_t)M * E_DIM * 2;
  __bf16* Kb      = (__bf16*)w; w += (size_t)M * E_DIM * 2;
  __bf16* Vb      = (__bf16*)w; w += (size_t)M * E_DIM * 2;
  __bf16* Ao      = (__bf16*)w; w += (size_t)M * E_DIM * 2;

  // 1) convert inputs to bf16 (weights transposed to [N][K])
  {
    int n = M * E_DIM;
    cvt_f32_to_bf16<<<(n + 255) / 256, 256, 0, stream>>>(x, x_bf, n);
    n = E_DIM * N3;
    cvt_transpose_bf16<<<(n + 255) / 256, 256, 0, stream>>>(w_qkv, wqkv_t, E_DIM, N3);
    n = E_DIM * E_DIM;
    cvt_transpose_bf16<<<(n + 255) / 256, 256, 0, stream>>>(w_proj, wproj_t, E_DIM, E_DIM);
  }

  // 2) QKV projection -> per-head Q/K/V
  gemm_bf16<1><<<dim3(N3 / 128, M / 128), 256, 0, stream>>>(
      x_bf, wqkv_t, b_qkv, nullptr, Qb, Kb, Vb, M, N3, E_DIM, T);

  // 3) flash attention
  flash_attn<<<(M / T) * N_HEAD * (T / 64), 128, 0, stream>>>(Qb, Kb, Vb, Ao, T);

  // 4) output projection -> f32 d_out
  gemm_bf16<0><<<dim3(E_DIM / 128, M / 128), 256, 0, stream>>>(
      Ao, wproj_t, b_proj, (float*)d_out, nullptr, nullptr, nullptr,
      M, E_DIM, E_DIM, T);
}
